// mAtt_bci_43018392437201
// MI455X (gfx1250) — compile-verified, matrix operand order checked by reading, not run on hardware
//
#include <hip/hip_runtime.h>

// ---------------------------------------------------------------------------
// mAtt-BCI pipeline for gfx1250 (MI455X, wave32, WMMA + TDM)
// conv1 -> BN+ELU -> conv2 (12 per-tap WMMA GEMMs, TDM-staged LDS window)
// -> BN+ELU -> Gram via WMMA -> Jacobi eigh -> MGS QR -> attention -> linear
// ---------------------------------------------------------------------------

typedef __attribute__((ext_vector_type(16))) _Float16 v16h;
typedef __attribute__((ext_vector_type(8)))  _Float16 v8h;
typedef __attribute__((ext_vector_type(8)))  float    v8f;
typedef __attribute__((ext_vector_type(4)))  unsigned v4u;
typedef __attribute__((ext_vector_type(8)))  int      v8i;
typedef __attribute__((ext_vector_type(4)))  int      v4i;

#define BS   256
#define CE   22
#define TT   1000
#define O2   96
#define KW   12
#define T2   989
#define T2P  992          // padded K for Gram (multiple of 32)
#define WIN  140          // conv2 LDS window: 8*16 + 12

#if defined(__has_builtin)
#  if __has_builtin(__builtin_amdgcn_tensor_load_to_lds)
#    define HAVE_TDM 1
#  endif
#endif

// ---------------- workspace layout (bytes, all 256B aligned) ----------------
constexpr size_t N_H1   = (size_t)BS * CE * TT;        // 5,632,000
constexpr size_t N_OUT2 = (size_t)BS * O2 * T2;        // 24,305,664
constexpr size_t N_XH   = (size_t)BS * O2 * T2P;       // 24,379,392
constexpr size_t OFF_H1   = 0;
constexpr size_t OFF_H1E  = OFF_H1  + N_H1 * 4;
constexpr size_t OFF_W2P  = OFF_H1E + N_H1 * 2;                 // 96*12*32 f16
constexpr size_t OFF_OUT2 = OFF_W2P + (size_t)O2 * KW * 32 * 2; // 73,728B
constexpr size_t OFF_XH   = OFF_OUT2 + N_OUT2 * 4;
constexpr size_t OFF_ST1  = OFF_XH + N_XH * 2;
constexpr size_t OFF_ST2  = OFF_ST1 + 256;
constexpr size_t OFF_S    = OFF_ST2 + 1024;
constexpr size_t OFF_QQT  = OFF_S   + (size_t)BS * 3 * 1024 * 4;
constexpr size_t OFF_KKT  = OFF_QQT + (size_t)BS * 3 * 256 * 4;
constexpr size_t OFF_VP   = OFF_KKT + (size_t)BS * 3 * 256 * 4;

__device__ __forceinline__ v16h cat8(v8h a, v8h b) {
  return __builtin_shufflevector(a, b, 0, 1, 2, 3, 4, 5, 6, 7,
                                 8, 9, 10, 11, 12, 13, 14, 15);
}

// ---------------------------------------------------------------------------
__global__ __launch_bounds__(256) void zero_stats(float* st1, float* st2) {
  int t = threadIdx.x;
  if (t < 2 * CE) st1[t] = 0.f;
  if (t < 2 * O2) st2[t] = 0.f;
}

// repack conv2 weights: w2p[(o*12 + k)*32 + c] = w2[o,c,k], c padded to 32
__global__ __launch_bounds__(256) void pack_w2(const float* __restrict__ w2,
                                               _Float16* __restrict__ w2p) {
  int i = blockIdx.x * 256 + threadIdx.x;       // over 96*12*32
  if (i >= O2 * KW * 32) return;
  int c = i & 31, ok = i >> 5;
  int o = ok / KW, k = ok % KW;
  w2p[i] = (c < CE) ? (_Float16)w2[o * (CE * KW) + c * KW + k] : (_Float16)0.f;
}

// ---------------- conv1: (256*1000, 22) x (22,22) + BN partial sums ---------
__global__ __launch_bounds__(256) void conv1_kernel(
    const float* __restrict__ x, const float* __restrict__ w1,
    const float* __restrict__ b1, float* __restrict__ h1,
    float* __restrict__ st1) {
  __shared__ float ws[CE * CE];
  __shared__ float wb[CE];
  __shared__ float ssum[CE], ssq[CE];
  int tid = threadIdx.x;
  for (int i = tid; i < CE * CE; i += 256) ws[i] = w1[i];
  if (tid < CE) { wb[tid] = b1[tid]; ssum[tid] = 0.f; ssq[tid] = 0.f; }
  __syncthreads();

  int g = blockIdx.x * 256 + tid;
  int b = g / TT, t = g % TT;
  const float* xb = x + (size_t)b * CE * TT + t;
  float xv[CE];
#pragma unroll
  for (int c = 0; c < CE; ++c) xv[c] = xb[c * TT];

  for (int o = 0; o < CE; ++o) {
    float acc = wb[o];
#pragma unroll
    for (int c = 0; c < CE; ++c) acc += xv[c] * ws[o * CE + c];
    h1[(size_t)b * CE * TT + o * TT + t] = acc;
    atomicAdd(&ssum[o], acc);
    atomicAdd(&ssq[o], acc * acc);
  }
  __syncthreads();
  if (tid < CE) {
    atomicAdd(&st1[tid * 2 + 0], ssum[tid]);
    atomicAdd(&st1[tid * 2 + 1], ssq[tid]);
  }
}

__global__ __launch_bounds__(256) void bn1_fin(
    const float* __restrict__ h1, const float* __restrict__ st1,
    const float* __restrict__ g, const float* __restrict__ bta,
    _Float16* __restrict__ h1e) {
  size_t i = (size_t)blockIdx.x * 256 + threadIdx.x;
  if (i >= N_H1) return;
  int o = (int)((i / TT) % CE);
  const float N = (float)BS * TT;
  float mu  = st1[o * 2] / N;
  float var = st1[o * 2 + 1] / N - mu * mu;
  float v = (h1[i] - mu) * rsqrtf(var + 1e-5f) * g[o] + bta[o];
  v = v > 0.f ? v : (expf(v) - 1.f);
  h1e[i] = (_Float16)v;
}

// ---------------- conv2: 12 per-tap 16x16x32 WMMA GEMMs --------------------
// Block: 256 threads = 8 waves; wave w owns t-tile (blockIdx.x*8 + w),
// computes all 6 o-tiles. Input window staged by TDM, transposed in LDS.
__global__ __launch_bounds__(256) void conv2_wmma(
    const _Float16* __restrict__ h1e, const _Float16* __restrict__ w2p,
    const float* __restrict__ b2, float* __restrict__ out2,
    float* __restrict__ st2) {
  __shared__ __align__(16) _Float16 ldsA[CE * WIN];   // [c][tw] row-major
  __shared__ __align__(16) _Float16 shT[WIN * 32];    // [tw][c pad 32]
  const int tid = threadIdx.x;
  const int b = blockIdx.z;
  const int t0blk = blockIdx.x * 128;
  const _Float16* src = h1e + (size_t)b * CE * TT;

#if HAVE_TDM
  if (tid < 32) {
    unsigned lds_off = (unsigned)(uintptr_t)(void*)ldsA;
    unsigned long long ga =
        (unsigned long long)(uintptr_t)(const void*)src + 2ull * (unsigned)t0blk;
    v4u g0;
    g0[0] = 1u;                                        // count=1
    g0[1] = lds_off;                                   // lds_addr
    g0[2] = (unsigned)(ga & 0xffffffffull);            // global_addr lo
    g0[3] = (unsigned)((ga >> 32) & 0x1ffffffull) | 0x80000000u; // hi | type=2
    v8i g1;
    g1[0] = (int)(1u << 16);                 // data_size=2B
    g1[1] = (int)(((unsigned)TT) << 16);     // tensor_dim0[15:0]=1000
    g1[2] = (int)(((unsigned)CE) << 16);     // dim0 hi=0, tensor_dim1=22
    g1[3] = (int)(((unsigned)WIN) << 16);    // tile_dim0=140
    g1[4] = (int)CE;                         // tile_dim1=22, tile_dim2=0
    g1[5] = (int)TT;                         // tensor_dim0_stride=1000
    g1[6] = 0; g1[7] = 0;
    v4i g2 = {0, 0, 0, 0};
    v4i g3 = {0, 0, 0, 0};
#if __clang_major__ >= 23
    v8i gz = {0, 0, 0, 0, 0, 0, 0, 0};
    __builtin_amdgcn_tensor_load_to_lds(g0, g1, g2, g3, gz, 0);
#else
    __builtin_amdgcn_tensor_load_to_lds(g0, g1, g2, g3, 0);
#endif
    __builtin_amdgcn_s_wait_tensorcnt(0);
  }
  __syncthreads();
#else
  for (int idx = tid; idx < CE * WIN; idx += 256) {
    int c = idx / WIN, tw = idx % WIN;
    int t = t0blk + tw;
    ldsA[idx] = (t < TT) ? src[c * TT + t] : (_Float16)0.f;
  }
  __syncthreads();
#endif

  // transpose to [tw][c], pad channels 22..31 with zero
  for (int idx = tid; idx < WIN * 32; idx += 256) {
    int tw = idx >> 5, c = idx & 31;
    shT[idx] = (c < CE) ? ldsA[c * WIN + tw] : (_Float16)0.f;
  }
  __syncthreads();

  const int lane = tid & 31;
  const int w    = tid >> 5;          // wave id = local t-tile
  const int m    = lane & 15;
  const int half = lane >> 4;
  const int koff = half * 8;

  v8f acc[6] = {};
#pragma unroll
  for (int k = 0; k < KW; ++k) {
    const v8h* pa = (const v8h*)(shT + (w * 16 + m + k) * 32 + koff);
    v16h a = cat8(pa[0], pa[2]);      // K-chunks [koff..+7], [koff+16..+23]
#pragma unroll
    for (int ot = 0; ot < 6; ++ot) {
      int o = ot * 16 + m;
      const v8h* pb = (const v8h*)(w2p + ((size_t)o * KW + k) * 32 + koff);
      v16h bf = cat8(pb[0], pb[2]);
      acc[ot] = __builtin_amdgcn_wmma_f32_16x16x32_f16(
          false, a, false, bf, (short)0, acc[ot], false, false);
    }
  }

#pragma unroll
  for (int ot = 0; ot < 6; ++ot) {
    int o = ot * 16 + m;
    float bias = b2[o];
    float s1 = 0.f, s2 = 0.f;
#pragma unroll
    for (int r = 0; r < 8; ++r) {
      int t = t0blk + w * 16 + half * 8 + r;
      if (t < T2) {
        float v = acc[ot][r] + bias;
        out2[((size_t)b * O2 + o) * T2 + t] = v;
        s1 += v; s2 += v * v;
      }
    }
    s1 += __shfl_xor(s1, 16, 32);
    s2 += __shfl_xor(s2, 16, 32);
    if (half == 0) {
      atomicAdd(&st2[o * 2 + 0], s1);
      atomicAdd(&st2[o * 2 + 1], s2);
    }
  }
}

// BN2 + ELU, writing f16 into T-padded X buffer for the Gram kernel
__global__ __launch_bounds__(256) void bn2_fin(
    const float* __restrict__ out2, const float* __restrict__ st2,
    const float* __restrict__ g, const float* __restrict__ bta,
    _Float16* __restrict__ Xh) {
  size_t i = (size_t)blockIdx.x * 256 + threadIdx.x;
  if (i >= N_OUT2) return;
  int ro = (int)(i / T2);
  int t  = (int)(i % T2);
  int o  = ro % O2;
  const float N = (float)BS * T2;
  float mu  = st2[o * 2] / N;
  float var = st2[o * 2 + 1] / N - mu * mu;
  float v = (out2[i] - mu) * rsqrtf(var + 1e-5f) * g[o] + bta[o];
  v = v > 0.f ? v : (expf(v) - 1.f);
  Xh[(size_t)ro * T2P + t] = (_Float16)v;
}

__global__ __launch_bounds__(256) void pad_xh(_Float16* __restrict__ Xh) {
  int i = blockIdx.x * 256 + threadIdx.x;       // BS*O2 rows * 3 pad cols
  if (i >= BS * O2 * (T2P - T2)) return;
  int row = i / (T2P - T2), j = i % (T2P - T2);
  Xh[(size_t)row * T2P + T2 + j] = (_Float16)0.f;
}

// ---------------- S = X X^T per (b,m): one wave does the full 32x32 ---------
__global__ __launch_bounds__(32) void s_wmma(const _Float16* __restrict__ Xh,
                                             float* __restrict__ S) {
  int lane = threadIdx.x, bm = blockIdx.x;
  int m = lane & 15, half = lane >> 4, koff = half * 8;
  const _Float16* r0 = Xh + ((size_t)bm * 32 + m) * T2P;
  const _Float16* r1 = r0 + 16 * T2P;
  v8f a00 = {}, a01 = {}, a10 = {}, a11 = {};
  for (int tt = 0; tt < T2P; tt += 32) {
    if ((tt & 127) == 0 && tt + 256 < T2P) {
      __builtin_prefetch(r0 + tt + 256, 0, 0);   // global_prefetch_b8
      __builtin_prefetch(r1 + tt + 256, 0, 0);
    }
    const v8h* p0 = (const v8h*)(r0 + tt + koff);
    const v8h* p1 = (const v8h*)(r1 + tt + koff);
    v16h f0 = cat8(p0[0], p0[2]);
    v16h f1 = cat8(p1[0], p1[2]);
    a00 = __builtin_amdgcn_wmma_f32_16x16x32_f16(false, f0, false, f0,
                                                 (short)0, a00, false, false);
    a01 = __builtin_amdgcn_wmma_f32_16x16x32_f16(false, f0, false, f1,
                                                 (short)0, a01, false, false);
    a10 = __builtin_amdgcn_wmma_f32_16x16x32_f16(false, f1, false, f0,
                                                 (short)0, a10, false, false);
    a11 = __builtin_amdgcn_wmma_f32_16x16x32_f16(false, f1, false, f1,
                                                 (short)0, a11, false, false);
  }
  float* Sb = S + (size_t)bm * 1024;
#pragma unroll
  for (int r = 0; r < 8; ++r) {
    int rr = half * 8 + r;
    Sb[rr * 32 + m]                  = a00[r];
    Sb[rr * 32 + 16 + m]             = a01[r];
    Sb[(16 + rr) * 32 + m]           = a10[r];
    Sb[(16 + rr) * 32 + 16 + m]      = a11[r];
  }
}

// ---------------- eigh (parallel Jacobi) + wq/wk/wv projection + MGS QR -----
__global__ __launch_bounds__(32) void eig_qr(
    const float* __restrict__ S, const float* __restrict__ wq,
    const float* __restrict__ wk, const float* __restrict__ wv,
    float* __restrict__ QQt, float* __restrict__ KKt, float* __restrict__ Vp) {
  __shared__ float A[32][33];
  __shared__ float Vm[32][33];
  __shared__ float cS[16], sS[16];
  __shared__ int   pS[16], qS[16];
  __shared__ float Up[32][8];
  __shared__ float Mx[16][8];
  __shared__ float Qx[16][8];
  __shared__ float red[32];
  __shared__ float dval[32];
  __shared__ int   sel[8];
  int lane = threadIdx.x;
  int bm = blockIdx.x;
  const float* Sb = S + (size_t)bm * 1024;
  for (int r = 0; r < 32; ++r) {
    A[r][lane]  = Sb[r * 32 + lane];
    Vm[r][lane] = (r == lane) ? 1.f : 0.f;
  }
  __syncthreads();

  for (int sweep = 0; sweep < 10; ++sweep)
    for (int rnd = 0; rnd < 31; ++rnd) {
      if (lane < 16) {
        int j1 = lane, j2 = 31 - lane;
        int p = (j1 == 0) ? 0 : (1 + ((j1 - 1 + rnd) % 31));
        int q = 1 + ((j2 - 1 + rnd) % 31);
        if (p > q) { int tw = p; p = q; q = tw; }
        float app = A[p][p], aqq = A[q][q], apq = A[p][q];
        float c = 1.f, s = 0.f;
        if (fabsf(apq) > 1e-12f) {
          float tau = (aqq - app) / (2.f * apq);
          float tr  = (tau >= 0.f ? 1.f : -1.f) /
                      (fabsf(tau) + sqrtf(1.f + tau * tau));
          c = rsqrtf(1.f + tr * tr); s = tr * c;
        }
        cS[lane] = c; sS[lane] = s; pS[lane] = p; qS[lane] = q;
      }
      __syncthreads();
      for (int k2 = 0; k2 < 16; ++k2) {       // rows: lane owns a column
        int p = pS[k2], q = qS[k2]; float c = cS[k2], s = sS[k2];
        float ap = A[p][lane], aq = A[q][lane];
        A[p][lane] = c * ap - s * aq;
        A[q][lane] = s * ap + c * aq;
      }
      __syncthreads();
      for (int k2 = 0; k2 < 16; ++k2) {       // cols: lane owns a row
        int p = pS[k2], q = qS[k2]; float c = cS[k2], s = sS[k2];
        float ap = A[lane][p], aq = A[lane][q];
        A[lane][p] = c * ap - s * aq;
        A[lane][q] = s * ap + c * aq;
        float vp2 = Vm[lane][p], vq = Vm[lane][q];
        Vm[lane][p] = c * vp2 - s * vq;
        Vm[lane][q] = s * vp2 + c * vq;
      }
      __syncthreads();
    }

  dval[lane] = A[lane][lane];
  __syncthreads();
  if (lane == 0) {                             // top-8 eigenvalues, descending
    unsigned used = 0;
    for (int j = 0; j < 8; ++j) {
      int best = 0; float bv = -3.4e38f;
      for (int i2 = 0; i2 < 32; ++i2)
        if (!((used >> i2) & 1) && dval[i2] > bv) { bv = dval[i2]; best = i2; }
      used |= 1u << best; sel[j] = best;
    }
  }
  __syncthreads();
  for (int j = 0; j < 8; ++j) Up[lane][j] = Vm[lane][sel[j]];
  __syncthreads();

  const float* Wt[3] = {wq, wk, wv};
  float* Ot[3] = {QQt + (size_t)bm * 256, KKt + (size_t)bm * 256,
                  Vp + (size_t)bm * 256};
  for (int wsel = 0; wsel < 3; ++wsel) {
    const float* W = Wt[wsel];
    if (lane < 16)
      for (int p2 = 0; p2 < 8; ++p2) {
        float acc = 0.f;
        for (int i2 = 0; i2 < 32; ++i2) acc += W[lane * 32 + i2] * Up[i2][p2];
        Mx[lane][p2] = acc;
      }
    __syncthreads();
    for (int j = 0; j < 8; ++j) {              // modified Gram-Schmidt
      red[lane] = (lane < 16) ? Mx[lane][j] * Mx[lane][j] : 0.f;
      __syncthreads();
      if (lane == 0) {
        float s2 = 0.f;
        for (int i2 = 0; i2 < 16; ++i2) s2 += red[i2];
        red[0] = rsqrtf(s2 + 1e-20f);
      }
      __syncthreads();
      float inv = red[0];
      if (lane < 16) Qx[lane][j] = Mx[lane][j] * inv;
      __syncthreads();
      for (int k2 = j + 1; k2 < 8; ++k2) {
        red[lane] = (lane < 16) ? Qx[lane][j] * Mx[lane][k2] : 0.f;
        __syncthreads();
        if (lane == 0) {
          float s2 = 0.f;
          for (int i2 = 0; i2 < 16; ++i2) s2 += red[i2];
          red[0] = s2;
        }
        __syncthreads();
        float d = red[0];
        if (lane < 16) Mx[lane][k2] -= d * Qx[lane][j];
        __syncthreads();
      }
    }
    if (lane < 16)
      for (int e = 0; e < 16; ++e) {
        float acc = 0.f;
        for (int p2 = 0; p2 < 8; ++p2) acc += Qx[lane][p2] * Qx[e][p2];
        Ot[wsel][lane * 16 + e] = acc;
      }
    __syncthreads();
  }
}

// ---------------- attention combine + linear head ---------------------------
__global__ __launch_bounds__(128) void attn_final(
    const float* __restrict__ QQt, const float* __restrict__ KKt,
    const float* __restrict__ Vp, const float* __restrict__ lin_w,
    const float* __restrict__ lin_b, float* __restrict__ out) {
  __shared__ float e2[9];
  __shared__ float wj[3];
  __shared__ float red[128];
  int b = blockIdx.x, tid = threadIdx.x;
  if (tid < 9) e2[tid] = 0.f;
  __syncthreads();
  for (int idx = tid; idx < 9 * 256; idx += 128) {
    int ij = idx >> 8, de = idx & 255;
    int i = ij / 3, j = ij % 3;
    int d = de >> 4, e = de & 15;
    const float* Qj = QQt + ((size_t)b * 3 + j) * 256;
    const float* Ki = KKt + ((size_t)b * 3 + i) * 256;
    float acc = 0.f;
    for (int k = 0; k < 16; ++k)
      acc += (Qj[d * 16 + k] - Ki[d * 16 + k]) *
             (Qj[e * 16 + k] - Ki[e * 16 + k]);
    atomicAdd(&e2[ij], acc * acc);
  }
  __syncthreads();
  if (tid < 3) {
    int j = tid;
    float s[3];
    for (int i = 0; i < 3; ++i) {
      float E = sqrtf(e2[i * 3 + j]);
      s[i] = 1.f / (1.f + log1pf(E));
    }
    float mx = fmaxf(s[0], fmaxf(s[1], s[2]));
    float den = expf(s[0] - mx) + expf(s[1] - mx) + expf(s[2] - mx);
    wj[j] = expf(s[2] - mx) / den;
  }
  __syncthreads();
  const float* Vl = Vp + ((size_t)b * 3 + 2) * 256;
  for (int u = 0; u < 4; ++u) {
    float part = 0.f;
    for (int idx = tid; idx < 768; idx += 128) {
      int j = idx >> 8, de = idx & 255;
      part += wj[j] * Vl[de] * lin_w[u * 768 + idx];
    }
    red[tid] = part; __syncthreads();
    for (int off = 64; off > 0; off >>= 1) {
      if (tid < off) red[tid] += red[tid + off];
      __syncthreads();
    }
    if (tid == 0) out[b * 4 + u] = red[0] + lin_b[u];
    __syncthreads();
  }
}

// ---------------------------------------------------------------------------
extern "C" void kernel_launch(void* const* d_in, const int* in_sizes, int n_in,
                              void* d_out, int out_size, void* d_ws,
                              size_t ws_size, hipStream_t stream) {
  const float* x      = (const float*)d_in[0];
  const float* w1     = (const float*)d_in[1];
  const float* b1     = (const float*)d_in[2];
  const float* g1     = (const float*)d_in[3];
  const float* bt1    = (const float*)d_in[4];
  const float* w2     = (const float*)d_in[5];
  const float* b2     = (const float*)d_in[6];
  const float* g2     = (const float*)d_in[7];
  const float* bt2    = (const float*)d_in[8];
  const float* wq     = (const float*)d_in[9];
  const float* wk     = (const float*)d_in[10];
  const float* wv     = (const float*)d_in[11];
  const float* lin_w  = (const float*)d_in[12];
  const float* lin_b  = (const float*)d_in[13];
  float* outp = (float*)d_out;

  char* ws = (char*)d_ws;
  float*    h1   = (float*)(ws + OFF_H1);
  _Float16* h1e  = (_Float16*)(ws + OFF_H1E);
  _Float16* w2p  = (_Float16*)(ws + OFF_W2P);
  float*    out2 = (float*)(ws + OFF_OUT2);
  _Float16* Xh   = (_Float16*)(ws + OFF_XH);
  float*    st1  = (float*)(ws + OFF_ST1);
  float*    st2  = (float*)(ws + OFF_ST2);
  float*    Sm   = (float*)(ws + OFF_S);
  float*    QQt  = (float*)(ws + OFF_QQT);
  float*    KKt  = (float*)(ws + OFF_KKT);
  float*    Vpm  = (float*)(ws + OFF_VP);

  zero_stats<<<1, 256, 0, stream>>>(st1, st2);
  pack_w2<<<(O2 * KW * 32 + 255) / 256, 256, 0, stream>>>(w2, w2p);
  conv1_kernel<<<(BS * TT) / 256, 256, 0, stream>>>(x, w1, b1, h1, st1);
  bn1_fin<<<(int)((N_H1 + 255) / 256), 256, 0, stream>>>(h1, st1, g1, bt1, h1e);
  conv2_wmma<<<dim3(8, 1, BS), 256, 0, stream>>>(h1e, w2p, b2, out2, st2);
  bn2_fin<<<(int)((N_OUT2 + 255) / 256), 256, 0, stream>>>(out2, st2, g2, bt2,
                                                           Xh);
  pad_xh<<<(BS * O2 * (T2P - T2) + 255) / 256, 256, 0, stream>>>(Xh);
  s_wmma<<<BS * 3, 32, 0, stream>>>(Xh, Sm);
  eig_qr<<<BS * 3, 32, 0, stream>>>(Sm, wq, wk, wv, QQt, KKt, Vpm);
  attn_final<<<BS, 128, 0, stream>>>(QQt, KKt, Vpm, lin_w, lin_b, outp);
}
